// SparseGPT_68384469287425
// MI455X (gfx1250) — matrix operand order repaired and checked by reading, not compiled, not run on hardware
//
#include <hip/hip_runtime.h>
#include <stdint.h>

// ---------------------------------------------------------------------------
// Problem constants (from reference)
// ---------------------------------------------------------------------------
#define T_SEQ 2048
#define NH    16
#define HS    128
#define NEA   2048
#define NEB   1024
#define NATT  2048
#define QKVN  6144   // 3 * NATT

typedef __attribute__((ext_vector_type(16))) __bf16 v16bf;
typedef __attribute__((ext_vector_type(8)))  float  v8f;
typedef unsigned int v4u __attribute__((ext_vector_type(4)));
typedef int          v4i __attribute__((ext_vector_type(4)));
typedef int          v8i __attribute__((ext_vector_type(8)));

struct Frag32B { uint4 lo, hi; };

static __device__ __forceinline__ v16bf frag_from(uint4 lo, uint4 hi) {
  Frag32B t; t.lo = lo; t.hi = hi;
  return __builtin_bit_cast(v16bf, t);
}

static __device__ __forceinline__ unsigned short f2bf(float f) {
  unsigned int u = __float_as_uint(f);
  unsigned int r = (u + 0x7FFFu + ((u >> 16) & 1u)) >> 16; // RNE
  return (unsigned short)r;
}
static __device__ __forceinline__ float bf2f(unsigned short h) {
  return __uint_as_float(((unsigned int)h) << 16);
}

// ---------------------------------------------------------------------------
// TDM: async DMA of a 2D bf16 tile (32 cols x 128 rows) global -> LDS,
// LDS rows padded to 40 elements (pad_interval=3: every 16 DWORDs,
// pad_amount=3: add 4 DWORDs). Tracked by TENSORcnt.
// This toolchain exposes the 6-arg builtin:
//   (v4u group0, v8i group1, v4i group2, v4i group3, v8i extra, i32 cpol)
// ---------------------------------------------------------------------------
static __device__ __forceinline__ void tdm_load_tile_2d(
    unsigned lds_off_bytes, const void* gptr, unsigned row_stride_elems)
{
  unsigned long long ga = (unsigned long long)(uintptr_t)gptr;
  v4u g0;
  g0[0] = 1u;                                   // count=1, user descriptor
  g0[1] = lds_off_bytes;                        // LDS byte address
  g0[2] = (unsigned)ga;                         // global addr [31:0]
  g0[3] = (unsigned)(ga >> 32) | 0x80000000u;   // addr[56:32] | type=2

  const unsigned tdim = 0x40000000u;            // huge dims: no OOB clipping
  v8i g1;
  g1[0] = (int)((1u << 16)      // data_size = 2 bytes
              | (1u << 20)      // pad_enable
              | (3u << 22)      // pad_interval: every 16 DWORDs (one row)
              | (3u << 25));    // pad_amount: 4 DWORDs (8 bf16)
  g1[1] = (int)((tdim & 0xFFFFu) << 16);               // tensor_dim0 lo16
  g1[2] = (int)((tdim >> 16) | ((tdim & 0xFFFFu) << 16)); // dim0 hi / dim1 lo
  g1[3] = (int)((tdim >> 16) | (32u << 16));           // dim1 hi | tile_dim0=32
  g1[4] = 128;                                         // tile_dim1=128 (tile_dim2=0)
  g1[5] = (int)row_stride_elems;                       // tensor_dim0_stride lo32
  g1[6] = 0;                                           // stride hi / dim1_stride lo
  g1[7] = 0;

  v4i gz;  gz[0] = 0; gz[1] = 0; gz[2] = 0; gz[3] = 0;   // groups 2/3: unused
  v8i gz8; gz8[0] = 0; gz8[1] = 0; gz8[2] = 0; gz8[3] = 0;
           gz8[4] = 0; gz8[5] = 0; gz8[6] = 0; gz8[7] = 0;
  __builtin_amdgcn_tensor_load_to_lds(g0, g1, gz, gz, gz8, 0);
}

// ---------------------------------------------------------------------------
// f32 -> bf16 copy / transpose
// ---------------------------------------------------------------------------
__global__ __launch_bounds__(256)
void cvt_f32_bf16(const float* __restrict__ s, unsigned short* __restrict__ d, int n) {
  int i = blockIdx.x * 256 + threadIdx.x;
  if (i < n) d[i] = f2bf(s[i]);
}

// src[K][N] (f32) -> dst[N][K] (bf16)
__global__ __launch_bounds__(256)
void tr_f32_bf16(const float* __restrict__ s, unsigned short* __restrict__ d, int Kd, int N) {
  int i = blockIdx.x * 256 + threadIdx.x;
  if (i < Kd * N) {
    int n = i % N, k = i / N;
    d[(size_t)n * Kd + k] = f2bf(s[i]);
  }
}

// ---------------------------------------------------------------------------
// Generic bf16 WMMA GEMM:  C[M][N] = A[M][K] * Bt[N][K]^T
// Block tile 128x128, K-step 32, 8 waves (2 m-waves x 4 n-waves), wave 64x32.
// LDS slabs are filled by double-buffered TDM DMAs issued by wave 0 and
// overlapped with WMMA on the other buffer.
// AMAP: row map applied to A reads   (0:id, 1: 2m, 2: 2m+1)  [stride-2 DMA]
// OMAP: row map applied to scatter   (1: 2m, 2: 2m+1) when OUTMODE==1
// OUTMODE: 0 = f32 row-major C, 1 = qkv scatter into Q/K/[V transposed]
// ---------------------------------------------------------------------------
template<int AMAP, int OMAP, int OUTMODE>
__global__ __launch_bounds__(256)
void gemm_bf16_kernel(const unsigned short* __restrict__ A,
                      const unsigned short* __restrict__ Bt,
                      float* __restrict__ C, int ldc,
                      unsigned short* __restrict__ Qp,
                      unsigned short* __restrict__ Kp,
                      unsigned short* __restrict__ Vp,
                      int Kd)
{
  __shared__ __align__(16) unsigned short As[2][128][40]; // [buf][m][k], padded
  __shared__ __align__(16) unsigned short Bs[2][128][40]; // [buf][n][k], padded

  const int tid  = threadIdx.x;
  const int wid  = tid >> 5, lane = tid & 31;
  const int half = lane >> 4, l16 = lane & 15;
  const int wm = wid & 1, wn = wid >> 1;
  const int m0 = blockIdx.y * 128, n0 = blockIdx.x * 128;

  v8f acc[4][2];
#pragma unroll
  for (int i = 0; i < 4; ++i)
#pragma unroll
    for (int j = 0; j < 2; ++j)
#pragma unroll
      for (int e = 0; e < 8; ++e) acc[i][j][e] = 0.f;

  // token row-map folds into a strided DMA
  const long arow0   = (AMAP == 0) ? (long)m0 : (2L * m0 + (AMAP == 2 ? 1 : 0));
  const unsigned sA  = (AMAP == 0) ? (unsigned)Kd : (unsigned)(2 * Kd);
  const unsigned short* Abase = A  + (size_t)arow0 * Kd;
  const unsigned short* Bbase = Bt + (size_t)n0 * Kd;

  const unsigned asOff = (unsigned)(size_t)&As[0][0][0];
  const unsigned bsOff = (unsigned)(size_t)&Bs[0][0][0];
  const unsigned bufBytes = 128u * 40u * 2u; // 10240

  if (wid == 0) {
    tdm_load_tile_2d(asOff, Abase, sA);
    tdm_load_tile_2d(bsOff, Bbase, (unsigned)Kd);
  }

  for (int kc = 0; kc < Kd; kc += 32) {
    const int buf = (kc >> 5) & 1;
    __builtin_amdgcn_s_wait_tensorcnt(0); // wave0: current tile done; others: no-op
    __syncthreads();                      // publish LDS to all waves
    if (wid == 0 && kc + 32 < Kd) {       // prefetch next tile into other buffer
      const int nb = buf ^ 1;
      tdm_load_tile_2d(asOff + nb * bufBytes, Abase + kc + 32, sA);
      tdm_load_tile_2d(bsOff + nb * bufBytes, Bbase + kc + 32, (unsigned)Kd);
    }

    // ---- fragments (ISA 16-bit A 16x32 / B 32x16 layouts) ----
    v16bf af[4], bfg[2];
#pragma unroll
    for (int ms = 0; ms < 4; ++ms) {
      int m = wm * 64 + ms * 16 + l16;
      int koff = half * 8;
      af[ms] = frag_from(*(const uint4*)&As[buf][m][koff],
                         *(const uint4*)&As[buf][m][koff + 16]);
    }
#pragma unroll
    for (int ns = 0; ns < 2; ++ns) {
      int n = wn * 32 + ns * 16 + l16;
      int kb = half * 8;
      bfg[ns] = frag_from(*(const uint4*)&Bs[buf][n][kb],
                          *(const uint4*)&Bs[buf][n][kb + 16]);
    }
#pragma unroll
    for (int ms = 0; ms < 4; ++ms)
#pragma unroll
      for (int ns = 0; ns < 2; ++ns)
        acc[ms][ns] = __builtin_amdgcn_wmma_f32_16x16x32_bf16(
            false, af[ms], false, bfg[ns], (short)0, acc[ms][ns], false, false);
  }

  // ---- epilogue ----
#pragma unroll
  for (int ms = 0; ms < 4; ++ms)
#pragma unroll
    for (int ns = 0; ns < 2; ++ns)
#pragma unroll
      for (int r = 0; r < 8; ++r) {
        int m = m0 + wm * 64 + ms * 16 + r + 8 * half; // C/D layout: M=r(+8), N=lane
        int n = n0 + wn * 32 + ns * 16 + l16;
        float v = acc[ms][ns][r];
        if (OUTMODE == 0) {
          C[(size_t)m * ldc + n] = v;
        } else {
          int g = (OMAP == 1) ? 2 * m : 2 * m + 1;     // interleave tokens
          int which = n >> 11;                          // 0:q 1:k 2:v
          int rr = n & (NATT - 1);
          int h = rr >> 7, d = rr & (HS - 1);
          int b = g >> 11, t = g & (T_SEQ - 1);
          int bh = b * NH + h;
          unsigned short bv = f2bf(v);
          if (which == 0)      Qp[((size_t)bh * T_SEQ + t) * HS + d] = bv;
          else if (which == 1) Kp[((size_t)bh * T_SEQ + t) * HS + d] = bv;
          else                 Vp[((size_t)bh * HS + d) * T_SEQ + t] = bv; // V^T
        }
      }
}

// ---------------------------------------------------------------------------
// RoPE (rotate-half) applied in place to bf16 Q and K [bh][t][128]
// ---------------------------------------------------------------------------
__global__ __launch_bounds__(256)
void rope_kernel(unsigned short* __restrict__ Q, unsigned short* __restrict__ K,
                 const float* __restrict__ cs, const float* __restrict__ sn)
{
  int i = blockIdx.x * 256 + threadIdx.x;      // over 32*2048*64
  int d  = i & 63;
  int t  = (i >> 6) & (T_SEQ - 1);
  int bh = i >> 17;
  if (bh >= 32) return;
  size_t base = ((size_t)bh * T_SEQ + t) * HS;
  float c1 = cs[t * HS + d],      s1 = sn[t * HS + d];
  float c2 = cs[t * HS + d + 64], s2 = sn[t * HS + d + 64];
  {
    float x1 = bf2f(Q[base + d]), x2 = bf2f(Q[base + d + 64]);
    Q[base + d]      = f2bf(x1 * c1 - x2 * s1);
    Q[base + d + 64] = f2bf(x2 * c2 + x1 * s2);
  }
  {
    float x1 = bf2f(K[base + d]), x2 = bf2f(K[base + d + 64]);
    K[base + d]      = f2bf(x1 * c1 - x2 * s1);
    K[base + d + 64] = f2bf(x2 * c2 + x1 * s2);
  }
}

// ---------------------------------------------------------------------------
// Causal flash attention.
// grid = (32 bh, 16 q-blocks), block = 256 (8 waves). Each wave: 16 q-rows.
// Q,K: [bh][t][128] bf16,  V: [bh][128][t] bf16 (transposed),
// Y out: [B*T][2048] bf16.
// ---------------------------------------------------------------------------
__global__ __launch_bounds__(256)
void attn_kernel(const unsigned short* __restrict__ Q,
                 const unsigned short* __restrict__ K,
                 const unsigned short* __restrict__ V,
                 unsigned short* __restrict__ Y)
{
  __shared__ __align__(16) unsigned short Pl[8][16][40]; // per-wave P scratch

  const int bh  = blockIdx.x;
  const int wid = threadIdx.x >> 5, lane = threadIdx.x & 31;
  const int half = lane >> 4, l16 = lane & 15;
  const int q0 = blockIdx.y * 128 + wid * 16;

  const unsigned short* Qb = Q + (size_t)bh * T_SEQ * HS;
  const unsigned short* Kb = K + (size_t)bh * T_SEQ * HS;
  const unsigned short* Vb = V + (size_t)bh * HS * T_SEQ;

  // Q A-fragments for this wave's 16 rows (K=128 -> 4 chunks of 32)
  v16bf qf[4];
  {
    const unsigned short* qrow = Qb + (size_t)(q0 + l16) * HS;
#pragma unroll
    for (int c = 0; c < 4; ++c) {
      int dbase = c * 32 + half * 8;
      qf[c] = frag_from(*(const uint4*)(qrow + dbase),
                        *(const uint4*)(qrow + dbase + 16));
    }
  }

  float mrow[8], lrow[8];
  v8f yacc[8];
#pragma unroll
  for (int r = 0; r < 8; ++r) { mrow[r] = -3.0e38f; lrow[r] = 0.f; }
#pragma unroll
  for (int dch = 0; dch < 8; ++dch)
#pragma unroll
    for (int e = 0; e < 8; ++e) yacc[dch][e] = 0.f;

  const float scale = 0.088388347648318447f; // 1/sqrt(128)
  const int nkt = (q0 + 15) / 32 + 1;        // causal: k-tiles of 32

  for (int kt = 0; kt < nkt; ++kt) {
    const int kb0 = kt * 32;

    // S = Q * K^T  (16 x 32 = two 16x16 accum frags)
    v8f s[2];
#pragma unroll
    for (int ns = 0; ns < 2; ++ns)
#pragma unroll
      for (int e = 0; e < 8; ++e) s[ns][e] = 0.f;
#pragma unroll
    for (int ns = 0; ns < 2; ++ns) {
      const unsigned short* krow = Kb + (size_t)(kb0 + ns * 16 + l16) * HS;
#pragma unroll
      for (int c = 0; c < 4; ++c) {
        int dbase = c * 32 + half * 8;
        v16bf kf = frag_from(*(const uint4*)(krow + dbase),
                             *(const uint4*)(krow + dbase + 16));
        s[ns] = __builtin_amdgcn_wmma_f32_16x16x32_bf16(
            false, qf[c], false, kf, (short)0, s[ns], false, false);
      }
    }

    // online softmax (row m = r + 8*half lives in 16 lanes of a half)
    float p[2][8];
#pragma unroll
    for (int r = 0; r < 8; ++r) {
      const int mg = q0 + r + 8 * half;
      float v0 = s[0][r] * scale;
      float v1 = s[1][r] * scale;
      if (kb0 + l16 > mg)      v0 = -3.0e38f;
      if (kb0 + 16 + l16 > mg) v1 = -3.0e38f;
      float mx = fmaxf(v0, v1);
#pragma unroll
      for (int off = 1; off < 16; off <<= 1)
        mx = fmaxf(mx, __shfl_xor(mx, off, 32));
      float nm   = fmaxf(mrow[r], mx);
      float corr = __expf(mrow[r] - nm);
      float e0 = __expf(v0 - nm);
      float e1 = __expf(v1 - nm);
      p[0][r] = e0; p[1][r] = e1;
      float ps = e0 + e1;
#pragma unroll
      for (int off = 1; off < 16; off <<= 1)
        ps += __shfl_xor(ps, off, 32);
      lrow[r] = lrow[r] * corr + ps;
      mrow[r] = nm;
#pragma unroll
      for (int dch = 0; dch < 8; ++dch) yacc[dch][r] *= corr;
    }

    // C-layout P -> A-layout via per-wave LDS (same-wave DS is in-order)
#pragma unroll
    for (int ns = 0; ns < 2; ++ns)
#pragma unroll
      for (int r = 0; r < 8; ++r)
        Pl[wid][r + 8 * half][ns * 16 + l16] = f2bf(p[ns][r]);
    __asm__ volatile("" ::: "memory");
    v16bf pf;
    {
      const unsigned short* prow = &Pl[wid][l16][0];
      int koff = half * 8;
      pf = frag_from(*(const uint4*)(prow + koff),
                     *(const uint4*)(prow + koff + 16));
    }

    // Y += P * V   (V^T rows are contiguous in k)
#pragma unroll
    for (int dch = 0; dch < 8; ++dch) {
      const unsigned short* vrow =
          Vb + (size_t)(dch * 16 + l16) * T_SEQ + kb0 + half * 8;
      v16bf vf = frag_from(*(const uint4*)(vrow),
                           *(const uint4*)(vrow + 16));
      yacc[dch] = __builtin_amdgcn_wmma_f32_16x16x32_bf16(
          false, pf, false, vf, (short)0, yacc[dch], false, false);
    }
  }

  // epilogue: Y[b*T+t][h*128+d] = yacc / l
  const int b = bh >> 4, h = bh & (NH - 1);
#pragma unroll
  for (int r = 0; r < 8; ++r) {
    int t = q0 + r + 8 * half;
    float inv = 1.0f / lrow[r];
    size_t rowbase = ((size_t)b * T_SEQ + t) * NATT + (size_t)h * HS;
#pragma unroll
    for (int dch = 0; dch < 8; ++dch)
      Y[rowbase + dch * 16 + l16] = f2bf(yacc[dch][r] * inv);
  }
}

// ---------------------------------------------------------------------------
// Host-side orchestration
// ---------------------------------------------------------------------------
extern "C" void kernel_launch(void* const* d_in, const int* in_sizes, int n_in,
                              void* d_out, int out_size, void* d_ws, size_t ws_size,
                              hipStream_t stream)
{
  (void)in_sizes; (void)n_in; (void)out_size; (void)ws_size;
  const float* x_a  = (const float*)d_in[0];
  const float* x_b  = (const float*)d_in[1];
  const float* cosb = (const float*)d_in[2];
  const float* sinb = (const float*)d_in[3];
  // d_in[4] = mask_a (fixed even/odd pattern; folded into index math)
  const float* Wa   = (const float*)d_in[5];
  const float* Wb   = (const float*)d_in[6];
  const float* Wpa  = (const float*)d_in[7];
  const float* Wpb  = (const float*)d_in[8];

  char* ws = (char*)d_ws;
  unsigned short* WaT  = (unsigned short*)(ws + 0);          // 6144x2048 bf16 (25.2 MB)
  unsigned short* Ybf  = (unsigned short*)(ws + 0);          // 4096x2048 bf16, reuses WaT
  unsigned short* WbT  = (unsigned short*)(ws + 25165824);   // 6144x1024
  unsigned short* xa16 = (unsigned short*)(ws + 37748736);   // 2048x2048
  unsigned short* xb16 = (unsigned short*)(ws + 46137344);   // 2048x1024
  unsigned short* WpaT = (unsigned short*)(ws + 50331648);   // 2048x2048
  unsigned short* WpbT = (unsigned short*)(ws + 58720256);   // 1024x2048
  unsigned short* Qbuf = (unsigned short*)(ws + 62914560);   // [32][2048][128]
  unsigned short* Kbuf = (unsigned short*)(ws + 79691776);   // [32][2048][128]
  unsigned short* Vbuf = (unsigned short*)(ws + 96468992);   // [32][128][2048]

  // precision conversions
  cvt_f32_bf16<<<(NEA * NEA) / 256, 256, 0, stream>>>(x_a, xa16, NEA * NEA);
  cvt_f32_bf16<<<(NEA * NEB) / 256, 256, 0, stream>>>(x_b, xb16, NEA * NEB);
  tr_f32_bf16<<<(NEA * QKVN) / 256, 256, 0, stream>>>(Wa,  WaT,  NEA, QKVN);
  tr_f32_bf16<<<(NEB * QKVN) / 256, 256, 0, stream>>>(Wb,  WbT,  NEB, QKVN);
  tr_f32_bf16<<<(NATT * NEA) / 256, 256, 0, stream>>>(Wpa, WpaT, NATT, NEA);
  tr_f32_bf16<<<(NATT * NEB) / 256, 256, 0, stream>>>(Wpb, WpbT, NATT, NEB);

  // QKV GEMMs with interleaving scatter (even tokens from a, odd from b)
  gemm_bf16_kernel<0, 1, 1><<<dim3(QKVN / 128, NEA / 128), 256, 0, stream>>>(
      xa16, WaT, nullptr, 0, Qbuf, Kbuf, Vbuf, NEA);
  gemm_bf16_kernel<0, 2, 1><<<dim3(QKVN / 128, NEA / 128), 256, 0, stream>>>(
      xb16, WbT, nullptr, 0, Qbuf, Kbuf, Vbuf, NEB);

  // RoPE on Q, K
  rope_kernel<<<(32 * T_SEQ * 64) / 256, 256, 0, stream>>>(Qbuf, Kbuf, cosb, sinb);

  // causal flash attention -> Ybf (aliases consumed WaT region)
  attn_kernel<<<dim3(32, T_SEQ / 128), 256, 0, stream>>>(Qbuf, Kbuf, Vbuf, Ybf);

  // projections: even rows -> y_a, odd rows -> y_b
  float* outA = (float*)d_out;
  float* outB = outA + (size_t)NEA * NEA;
  gemm_bf16_kernel<1, 0, 0><<<dim3(NEA / 128, NEA / 128), 256, 0, stream>>>(
      Ybf, WpaT, outA, NEA, nullptr, nullptr, nullptr, NATT);
  gemm_bf16_kernel<2, 0, 0><<<dim3(NEB / 128, NEA / 128), 256, 0, stream>>>(
      Ybf, WpbT, outB, NEB, nullptr, nullptr, nullptr, NATT);
}
